// MultiHeadAttention_NoMask_13142599926295
// MI455X (gfx1250) — matrix operand order repaired and checked by reading, not compiled, not run on hardware
//
#include <hip/hip_runtime.h>
#include <hip/hip_bf16.h>
#include <stdint.h>

// ---------------------------------------------------------------------------
// MHA forward for B=4, L=2048, D=1024, H=16, Dk=Dv=64 on gfx1250 (MI455X).
// GEMMs: v_wmma_f32_16x16x32_f16; tiles staged in LDS (double-buffered) with
// global_load_async_to_lds_b128 / ASYNCcnt; wave computes 32x64 (8 WMMAs/step).
// ---------------------------------------------------------------------------

typedef __attribute__((ext_vector_type(16))) _Float16 v16h;
typedef __attribute__((ext_vector_type(8)))  _Float16 v8h;
typedef __attribute__((ext_vector_type(8)))  float    v8f;

static constexpr int  Bc  = 4;
static constexpr int  Lc  = 2048;
static constexpr int  Dc  = 1024;
static constexpr int  Hc  = 16;
static constexpr long MR  = (long)Bc * Lc;        // 8192
static constexpr long NX  = MR * Dc;              // 8,388,608
static constexpr long NW  = (long)Dc * Dc;        // 1,048,576
static constexpr long ATTN_OFF = NX;

static constexpr int PA = 40;                     // padded LDS row pitch (halves)

// ---------------------------------------------------------------------------
__global__ void cvt_f32_to_f16(const float* __restrict__ in, _Float16* __restrict__ out, long n) {
    long i = (long)blockIdx.x * blockDim.x + threadIdx.x;
    if (i < n) out[i] = (_Float16)in[i];
}

// Vh [B*L, H*64] (f16)  ->  Vt [(b*H+h), 64, 2048] (f16) so attn@V is A*B^T form.
__global__ void transpose_v(const _Float16* __restrict__ src, _Float16* __restrict__ dst) {
    long i = (long)blockIdx.x * blockDim.x + threadIdx.x;   // i < 8192*1024
    int  c = (int)(i & (Dc - 1));
    long r = i >> 10;
    int  b = (int)(r >> 11);
    int  m = (int)(r & (Lc - 1));
    int  h = c >> 6;
    int  kk = c & 63;
    dst[((((long)b * Hc + h) * 64 + kk) << 11) + m] = src[i];
}

// ---------------------------------------------------------------------------
// Batched WMMA GEMM:  C[z] = alpha * A[z] * B[z]^T + bias (+ resid)
//   A: [M, K]  (AT = _Float16: staged in LDS; AT = float: direct global + cvt)
//   B: [N, K]  row-major f16, staged in LDS (8-way reuse across waves)
//   Block = 256 threads = 8 waves -> 256x64 tile; wave -> 32x64 strip
//   (2 A frags x 4 B frags = 8 WMMAs per 32-wide K step, double-buffered).
// ---------------------------------------------------------------------------
template <typename AT>
__global__ void gemm_wmma_f16(
    const AT* __restrict__ A, int lda, long sAb, long sAh,
    const _Float16* __restrict__ Bm, int ldb, long sBb, long sBh,
    float* __restrict__ Cf, _Float16* __restrict__ Ch, int ldc, long sCb, long sCh,
    const float* __restrict__ bias, const float* __restrict__ resid, int ldr,
    int Kdim, float alpha, int nBatch)
{
    constexpr bool A16 = (sizeof(AT) == 2);
    constexpr int  APT = A16 ? 5 : 1;   // async instructions per wave per tile

    __shared__ _Float16 sA[A16 ? (2 * 256 * PA) : 2];
    __shared__ _Float16 sB[2 * 64 * PA];

    const int z = blockIdx.z;
    const int b = z % nBatch;
    const int h = z / nBatch;

    const AT*       Ab = A  + (long)b * sAb + (long)h * sAh;
    const _Float16* Bb = Bm + (long)b * sBb + (long)h * sBh;
    const long    cOff =      (long)b * sCb + (long)h * sCh;

    const int tid      = threadIdx.x;
    const int wave     = tid >> 5;
    const int lane     = tid & 31;
    const int blockRow = blockIdx.y * 256;
    const int colBase  = blockIdx.x * 64;
    const int rowBase  = blockRow + wave * 32;      // 32-row strip per wave

    // fragment lane mapping (CDNA5 16-bit WMMA layouts)
    const int m   = lane & 15;
    const int kbA = (lane < 16) ? 0 : 8;    // A: k-base 0/8 per half-wave
    const int n   = lane & 15;
    const int kbB = (lane < 16) ? 0 : 8;    // B: k-base 0/8 per half-wave

    // async-copy chunk mapping: thread -> (row, 16B k-chunk)
    const int crow = tid >> 2;              // 0..63
    const int ckc  = (tid & 3) * 8;         // 0,8,16,24 (halves)

    const int nk = Kdim / 32;

    auto issueTile = [&](int kt, int buf) {
        const int k0 = kt * 32;
        if constexpr (A16) {
#pragma unroll
            for (int i = 0; i < 4; ++i) {   // A rows i*64 + crow, 256 rows total
                const int row = i * 64 + crow;
                const uint64_t g = (uint64_t)(uintptr_t)(Ab + (long)(blockRow + row) * lda + k0 + ckc);
                const uint32_t l = (uint32_t)(uintptr_t)&sA[buf * 256 * PA + row * PA + ckc];
                asm volatile("global_load_async_to_lds_b128 %0, %1, off" :: "v"(l), "v"(g) : "memory");
            }
        }
        {   // B rows 0..63
            const uint64_t g = (uint64_t)(uintptr_t)(Bb + (long)(colBase + crow) * ldb + k0 + ckc);
            const uint32_t l = (uint32_t)(uintptr_t)&sB[buf * 64 * PA + crow * PA + ckc];
            asm volatile("global_load_async_to_lds_b128 %0, %1, off" :: "v"(l), "v"(g) : "memory");
        }
    };

    // float-A path: direct global pointers for the two 16-row fragments
    const AT* aptr0 = Ab + (long)(rowBase + m) * lda + kbA;
    const AT* aptr1 = Ab + (long)(rowBase + 16 + m) * lda + kbA;

    v8f acc[2][4] = {{v8f{}, v8f{}, v8f{}, v8f{}}, {v8f{}, v8f{}, v8f{}, v8f{}}};

    issueTile(0, 0);
    for (int kt = 0; kt < nk; ++kt) {
        const int buf = kt & 1;
        __syncthreads();                          // everyone done reading buf^1 (previous iter)
        if (kt + 1 < nk) {
            issueTile(kt + 1, buf ^ 1);
            asm volatile("s_wait_asynccnt %0" :: "n"(APT) : "memory");  // tile kt landed
        } else {
            asm volatile("s_wait_asynccnt 0" ::: "memory");
        }
        __syncthreads();                          // tile kt visible to all waves

        // ---- preload ALL fragments of this k-step, then burst the WMMAs ----
        union F { v16h v; v8h h2[2]; _Float16 e[16]; };
        F af0, af1, bf[4];

        if constexpr (A16) {
            const long sa = (long)buf * 256 * PA;
            const int  ar0 = wave * 32 + m;
            af0.h2[0] = *(const v8h*)&sA[sa + ar0 * PA + kbA];
            af0.h2[1] = *(const v8h*)&sA[sa + ar0 * PA + kbA + 16];
            const int  ar1 = ar0 + 16;
            af1.h2[0] = *(const v8h*)&sA[sa + ar1 * PA + kbA];
            af1.h2[1] = *(const v8h*)&sA[sa + ar1 * PA + kbA + 16];
        } else {
            const int k0 = kt * 32;
            if (k0 + 32 < Kdim) {
                __builtin_prefetch(aptr0 + k0 + 32, 0, 1);
                __builtin_prefetch(aptr1 + k0 + 32, 0, 1);
            }
            v8f f0 = *(const v8f*)(aptr0 + k0);
            v8f f1 = *(const v8f*)(aptr0 + k0 + 16);
            v8f f2 = *(const v8f*)(aptr1 + k0);
            v8f f3 = *(const v8f*)(aptr1 + k0 + 16);
#pragma unroll
            for (int i = 0; i < 8; ++i) {
                af0.e[i]     = (_Float16)f0[i];
                af0.e[8 + i] = (_Float16)f1[i];
                af1.e[i]     = (_Float16)f2[i];
                af1.e[8 + i] = (_Float16)f3[i];
            }
        }
        {
            const long sb = (long)buf * 64 * PA;
#pragma unroll
            for (int j = 0; j < 4; ++j) {
                const int br = j * 16 + n;
                bf[j].h2[0] = *(const v8h*)&sB[sb + br * PA + kbB];
                bf[j].h2[1] = *(const v8h*)&sB[sb + br * PA + kbB + 16];
            }
        }
#pragma unroll
        for (int j = 0; j < 4; ++j)
            acc[0][j] = __builtin_amdgcn_wmma_f32_16x16x32_f16(
                            false, af0.v, false, bf[j].v, (short)0, acc[0][j], false, false);
#pragma unroll
        for (int j = 0; j < 4; ++j)
            acc[1][j] = __builtin_amdgcn_wmma_f32_16x16x32_f16(
                            false, af1.v, false, bf[j].v, (short)0, acc[1][j], false, false);
    }

    // C/D f32 layout: VGPR r holds row (r + 8*(lane>=16)), col = lane%16.
    const int rOff = (lane < 16) ? 0 : 8;
#pragma unroll
    for (int s = 0; s < 2; ++s) {
#pragma unroll
        for (int j = 0; j < 4; ++j) {
            const int col = colBase + j * 16 + n;
            const float bb = bias ? bias[col] : 0.0f;
            v8f a = acc[s][j];
#pragma unroll
            for (int r = 0; r < 8; ++r) {
                const int row = rowBase + s * 16 + rOff + r;
                float val = a[r] * alpha + bb;
                if (resid) val += resid[(long)row * ldr + col];
                const long ci = cOff + (long)row * ldc + col;
                if (Cf) Cf[ci] = val;
                if (Ch) Ch[ci] = (_Float16)val;
            }
        }
    }
}

// ---------------------------------------------------------------------------
// Row softmax over L=2048 with diagonal self-exclusion, in place on fp32 attn.
// ---------------------------------------------------------------------------
__global__ void softmax_rows(float* __restrict__ attn) {
    __shared__ float red[256];
    const long row = blockIdx.x;               // (h*B+b)*L + l
    const int  l   = (int)(row & (Lc - 1));
    float* p = attn + row * (long)Lc;
    const int tid = threadIdx.x;

    float vals[8];
    float vmax = -__builtin_inff();
#pragma unroll
    for (int i = 0; i < 8; ++i) {
        const int mcol = tid + i * 256;
        float x = p[mcol];
        if (mcol == l) x = -__builtin_inff();
        vals[i] = x;
        vmax = fmaxf(vmax, x);
    }
    red[tid] = vmax; __syncthreads();
    for (int s = 128; s > 0; s >>= 1) {
        if (tid < s) red[tid] = fmaxf(red[tid], red[tid + s]);
        __syncthreads();
    }
    vmax = red[0]; __syncthreads();

    float sum = 0.0f;
#pragma unroll
    for (int i = 0; i < 8; ++i) {
        float e = (vals[i] == -__builtin_inff()) ? 0.0f : __expf(vals[i] - vmax);
        vals[i] = e;
        sum += e;
    }
    red[tid] = sum; __syncthreads();
    for (int s = 128; s > 0; s >>= 1) {
        if (tid < s) red[tid] += red[tid + s];
        __syncthreads();
    }
    const float inv = 1.0f / red[0];
#pragma unroll
    for (int i = 0; i < 8; ++i) p[tid + i * 256] = vals[i] * inv;
}

// ---------------------------------------------------------------------------
// LayerNorm over D=1024 (residual already fused into x by the proj GEMM).
// ---------------------------------------------------------------------------
__global__ void layernorm_rows(const float* __restrict__ x,
                               const float* __restrict__ gamma,
                               const float* __restrict__ beta,
                               float* __restrict__ out) {
    __shared__ float red[256];
    const long row = blockIdx.x;
    const float* p = x + row * (long)Dc;
    const int tid = threadIdx.x;

    float v[4], s = 0.0f, s2 = 0.0f;
#pragma unroll
    for (int i = 0; i < 4; ++i) {
        v[i] = p[tid + i * 256];
        s  += v[i];
        s2 += v[i] * v[i];
    }
    red[tid] = s; __syncthreads();
    for (int st = 128; st > 0; st >>= 1) { if (tid < st) red[tid] += red[tid + st]; __syncthreads(); }
    const float mu = red[0] * (1.0f / Dc); __syncthreads();
    red[tid] = s2; __syncthreads();
    for (int st = 128; st > 0; st >>= 1) { if (tid < st) red[tid] += red[tid + st]; __syncthreads(); }
    const float var = red[0] * (1.0f / Dc) - mu * mu;
    const float rs  = rsqrtf(var + 1e-5f);
#pragma unroll
    for (int i = 0; i < 4; ++i) {
        const int c = tid + i * 256;
        out[row * (long)Dc + c] = (v[i] - mu) * rs * gamma[c] + beta[c];
    }
}

// ---------------------------------------------------------------------------
extern "C" void kernel_launch(void* const* d_in, const int* in_sizes, int n_in,
                              void* d_out, int out_size, void* d_ws, size_t ws_size,
                              hipStream_t stream) {
    (void)in_sizes; (void)n_in; (void)out_size; (void)ws_size;

    const float* q     = (const float*)d_in[0];
    const float* k     = (const float*)d_in[1];
    const float* v     = (const float*)d_in[2];
    const float* Wq    = (const float*)d_in[3];
    const float* bq    = (const float*)d_in[4];
    const float* Wk    = (const float*)d_in[5];
    const float* bk    = (const float*)d_in[6];
    const float* Wv    = (const float*)d_in[7];
    const float* bv    = (const float*)d_in[8];
    const float* Wo    = (const float*)d_in[9];
    const float* bo    = (const float*)d_in[10];
    const float* gamma = (const float*)d_in[11];
    const float* beta  = (const float*)d_in[12];

    float* outp = (float*)d_out;            // [B*L, D] normalized output
    float* attn = outp + ATTN_OFF;          // [(h*B+b), L, L] fp32 probabilities

    _Float16* Xq16 = (_Float16*)d_ws;
    _Float16* Xk16 = Xq16 + NX;
    _Float16* Xv16 = Xk16 + NX;
    _Float16* Wq16 = Xv16 + NX;
    _Float16* Wk16 = Wq16 + NW;
    _Float16* Wv16 = Wk16 + NW;
    _Float16* Wo16 = Wv16 + NW;
    _Float16* Qh   = Wo16 + NW;             // [B*L, H*64]
    _Float16* Kh   = Qh   + NX;
    _Float16* Vh   = Kh   + NX;
    _Float16* Vt   = Vh   + NX;             // [(b*H+h), 64, 2048]
    _Float16* Ctx  = Vt   + NX;             // [B*L, H*64]
    float*    Proj = (float*)(Ctx + NX);    // [B*L, D] fp32 (proj + residual)

    const dim3 blk(256);

    // 1) fp32 -> f16 conversions
    cvt_f32_to_f16<<<(int)(NX / 256), blk, 0, stream>>>(q,  Xq16, NX);
    cvt_f32_to_f16<<<(int)(NX / 256), blk, 0, stream>>>(k,  Xk16, NX);
    cvt_f32_to_f16<<<(int)(NX / 256), blk, 0, stream>>>(v,  Xv16, NX);
    cvt_f32_to_f16<<<(int)(NW / 256), blk, 0, stream>>>(Wq, Wq16, NW);
    cvt_f32_to_f16<<<(int)(NW / 256), blk, 0, stream>>>(Wk, Wk16, NW);
    cvt_f32_to_f16<<<(int)(NW / 256), blk, 0, stream>>>(Wv, Wv16, NW);
    cvt_f32_to_f16<<<(int)(NW / 256), blk, 0, stream>>>(Wo, Wo16, NW);

    // 2) QKV projections: [8192,1024] = X * W^T + b   (block tile 256x64)
    const dim3 gProj(Dc / 64, (int)(MR / 256), 1);
    gemm_wmma_f16<_Float16><<<gProj, blk, 0, stream>>>(
        Xq16, Dc, 0, 0, Wq16, Dc, 0, 0,
        nullptr, Qh, Dc, 0, 0, bq, nullptr, 0, Dc, 1.0f, 1);
    gemm_wmma_f16<_Float16><<<gProj, blk, 0, stream>>>(
        Xk16, Dc, 0, 0, Wk16, Dc, 0, 0,
        nullptr, Kh, Dc, 0, 0, bk, nullptr, 0, Dc, 1.0f, 1);
    gemm_wmma_f16<_Float16><<<gProj, blk, 0, stream>>>(
        Xv16, Dc, 0, 0, Wv16, Dc, 0, 0,
        nullptr, Vh, Dc, 0, 0, bv, nullptr, 0, Dc, 1.0f, 1);

    // 3) V transpose for attn@V
    transpose_v<<<(int)(NX / 256), blk, 0, stream>>>(Vh, Vt);

    // 4) scores: per (h,b): S = (Q_h K_h^T) / 8  -> fp32 into d_out attn region
    const dim3 gScore(Lc / 64, Lc / 256, Hc * Bc);
    gemm_wmma_f16<_Float16><<<gScore, blk, 0, stream>>>(
        Qh, Dc, (long)Lc * Dc, 64,
        Kh, Dc, (long)Lc * Dc, 64,
        attn, nullptr, Lc, (long)Lc * Lc, (long)Bc * Lc * Lc,
        nullptr, nullptr, 0, 64, 0.125f, Bc);

    // 5) softmax with diagonal exclusion, in place
    softmax_rows<<<Hc * Bc * Lc, blk, 0, stream>>>(attn);

    // 6) context: per (h,b): Ctx = attn(f32, cvt in-reg) * Vt^T  -> f16 [B*L, H*64]
    const dim3 gCtx(1, Lc / 256, Hc * Bc);
    gemm_wmma_f16<float><<<gCtx, blk, 0, stream>>>(
        attn, Lc, (long)Lc * Lc, (long)Bc * Lc * Lc,
        Vt, Lc, (long)Hc * 64 * Lc, (long)64 * Lc,
        nullptr, Ctx, Dc, (long)Lc * Dc, 64,
        nullptr, nullptr, 0, Lc, 1.0f, Bc);

    // 7) output projection + residual: Proj = Ctx * Wo^T + bo + q
    gemm_wmma_f16<_Float16><<<gProj, blk, 0, stream>>>(
        Ctx, Dc, 0, 0, Wo16, Dc, 0, 0,
        Proj, nullptr, Dc, 0, 0, bo, q, Dc, Dc, 1.0f, 1);

    // 8) LayerNorm -> d_out
    layernorm_rows<<<(int)MR, blk, 0, stream>>>(Proj, gamma, beta, outp);
}
